// OTAligner_56702158241900
// MI455X (gfx1250) — matrix operand order repaired and checked by reading, not compiled, not run on hardware
//
#include <hip/hip_runtime.h>
#include <hip/hip_bf16.h>

// ---------------------------------------------------------------------------
// OTAligner fused kernel for gfx1250 (MI455X).
//
// Sinkhorn with N=1 collapses; workload is:
//   z_m = x_m @ W_m^T   (three 4096x4096 @ 4096x128 GEMMs, fp32)
//   aligned = (z_ct + z_txt + z_lab)/3
//   ot_loss = sum[(zc-zt)^2+(zc-zl)^2+(zt-zl)^2] / (3*B) - eps
//
// fp32 operands split into bf16 hi + lo residual; 3 x v_wmma_f32_16x16x32_bf16
// per 16x16x32 tile (hi*hi + hi*lo + lo*hi) -> ~fp32 accuracy at bf16 rates.
//   * W pre-converted once to bf16 hi/lo in workspace (L2-resident, reused 256x).
//   * A tiles converted cooperatively into double-buffered LDS.
//   * Software pipelining: B fragments register-double-buffered; A path is
//     triple-staged (global->regs, regs->LDS, LDS->WMMA).
//   * Loop unrolled 2x so the current/next buffers swap by register renaming
//     (kills the v_dual_mov + v_nop hazard padding seen in round 3).
// ---------------------------------------------------------------------------

typedef __attribute__((ext_vector_type(16))) __bf16 v16bf;
typedef __attribute__((ext_vector_type(8)))  __bf16 v8bf;
typedef __attribute__((ext_vector_type(2)))  __bf16 v2bf;
typedef __attribute__((ext_vector_type(8)))  float  v8f;

#define KDIM   4096
#define NDIM   128
#define BATCH_ 4096
#define EPS_   0.05f
#define NW     (NDIM * KDIM)      // elements per W modality (524288)
#define NK     (KDIM / 32)        // k-steps (128, even -> exact 2x unroll)

// LDS tile: 16 rows x 32 bf16, row stride padded to 80 B (16B-aligned b128
// loads; 20-bank stride -> conflict-free across the 16 row readers).
#define ROWB   80
#define TILEB  (16 * ROWB)        // 1280 B (one hi or lo tile)
#define MODB   (2 * TILEB)        // hi + lo per modality
#define BUFB   (3 * MODB)         // 3 modalities per buffer
#define SMEMB  (2 * BUFB)         // double buffered: 15360 B

// --- one-time W fp32 -> bf16 hi/lo conversion ------------------------------
__global__ __launch_bounds__(256) void
convert_weights(const float* __restrict__ W0, const float* __restrict__ W1,
                const float* __restrict__ W2, __bf16* __restrict__ hi,
                __bf16* __restrict__ lo) {
  const int mIdx = blockIdx.y;
  const float* __restrict__ W = (mIdx == 0) ? W0 : (mIdx == 1) ? W1 : W2;
  __bf16* __restrict__ h = hi + (size_t)mIdx * NW;
  __bf16* __restrict__ l = lo + (size_t)mIdx * NW;
  for (int i = blockIdx.x * blockDim.x + threadIdx.x; i < NW;
       i += gridDim.x * blockDim.x) {
    const float x = W[i];
    const __bf16 hh = (__bf16)x;
    h[i] = hh;
    l[i] = (__bf16)(x - (float)hh);
  }
}

// --- A staging: split into load (iter n-2) and convert+store (iter n-1) ----
__device__ __forceinline__ float2 stage_load(const float* __restrict__ x,
                                             int m0, int k, int tid) {
  const int row = tid >> 4;       // 0..15
  const int cp  = tid & 15;       // column pair -> cols 2cp, 2cp+1
  const float* src = x + (size_t)(m0 + row) * KDIM + k + 2 * cp;
  __builtin_prefetch(src + 512, 0, 3);   // stream-ahead (2 KB) into GL2
  return *(const float2*)src;
}

__device__ __forceinline__ void stage_store(float2 v, char* smem, int base,
                                            int tid) {
  const int row = tid >> 4;
  const int cp  = tid & 15;
  const __bf16 h0 = (__bf16)v.x;
  const __bf16 h1 = (__bf16)v.y;
  const v2bf hp = {h0, h1};
  const v2bf lp = {(__bf16)(v.x - (float)h0), (__bf16)(v.y - (float)h1)};
  const int off = base + row * ROWB + cp * 4;
  *(v2bf*)(smem + off) = hp;
  *(v2bf*)(smem + off + TILEB) = lp;
}

// A fragment (16-bit A 16x32 wave32 layout): lane (m = lane&15, hv = lane>>4)
// holds K = hv*8+0..7 then 16+hv*8+0..7 -> two aligned 16 B LDS chunks.
__device__ __forceinline__ v16bf load_afrag(const char* smem, int base,
                                            int r16, int hv) {
  const int o = base + r16 * ROWB + hv * 16;
  const v8bf f0 = *(const v8bf*)(smem + o);
  const v8bf f1 = *(const v8bf*)(smem + o + 32);
  return __builtin_shufflevector(f0, f1, 0, 1, 2, 3, 4, 5, 6, 7, 8, 9, 10, 11,
                                 12, 13, 14, 15);
}

// B fragment (16-bit B 32x16 wave32 layout): lane (n = lane&15, hv = lane>>4)
// holds K = k + hv*8 + 0..7 then k + 16 + hv*8 + 0..7 -> two global b128 loads of bf16.
__device__ __forceinline__ v16bf load_bfrag(const __bf16* __restrict__ Wrow,
                                            int k, int hv) {
  const v8bf* p = (const v8bf*)(Wrow + k + hv * 8);
  const v8bf f0 = p[0];
  const v8bf f1 = p[2];
  return __builtin_shufflevector(f0, f1, 0, 1, 2, 3, 4, 5, 6, 7, 8, 9, 10, 11,
                                 12, 13, 14, 15);
}

__device__ __forceinline__ v8f wmma3(const v16bf& ahi, const v16bf& alo,
                                     const v16bf& bhi, const v16bf& blo,
                                     v8f acc) {
  acc = __builtin_amdgcn_wmma_f32_16x16x32_bf16(false, ahi, false, bhi,
                                                (short)0, acc, false, false);
  acc = __builtin_amdgcn_wmma_f32_16x16x32_bf16(false, ahi, false, blo,
                                                (short)0, acc, false, false);
  acc = __builtin_amdgcn_wmma_f32_16x16x32_bf16(false, alo, false, bhi,
                                                (short)0, acc, false, false);
  return acc;
}

// One block = 16 batch rows; 8 waves cover the 128 output columns.
__global__ __launch_bounds__(256) void
ot_fused_gemm(const float* __restrict__ x_ct, const float* __restrict__ x_txt,
              const float* __restrict__ x_lab, const __bf16* __restrict__ Whi,
              const __bf16* __restrict__ Wlo, float* __restrict__ aligned,
              float* __restrict__ partials) {
  __shared__ __align__(16) char smem[SMEMB];

  const int tid  = threadIdx.x;
  const int lane = tid & 31;
  const int wave = tid >> 5;
  const int r16  = lane & 15;
  const int hv   = lane >> 4;
  const int m0   = blockIdx.x * 16;
  const int n    = wave * 16 + r16;

  const __bf16* __restrict__ bh0 = Whi + (size_t)n * KDIM;
  const __bf16* __restrict__ bl0 = Wlo + (size_t)n * KDIM;
  const __bf16* __restrict__ bh1 = bh0 + NW;
  const __bf16* __restrict__ bl1 = bl0 + NW;
  const __bf16* __restrict__ bh2 = bh0 + 2 * (size_t)NW;
  const __bf16* __restrict__ bl2 = bl0 + 2 * (size_t)NW;

  v8f acc0 = {}, acc1 = {}, acc2 = {};

  // ---- pipeline prologue ----
  // chunk 0 -> regs -> LDS buf0; chunk 1 -> regs; B chunk 0 -> regs.
  float2 a0r = stage_load(x_ct,  m0, 0, tid);
  float2 a1r = stage_load(x_txt, m0, 0, tid);
  float2 a2r = stage_load(x_lab, m0, 0, tid);
  stage_store(a0r, smem, 0 * MODB, tid);
  stage_store(a1r, smem, 1 * MODB, tid);
  stage_store(a2r, smem, 2 * MODB, tid);
  a0r = stage_load(x_ct,  m0, 32, tid);
  a1r = stage_load(x_txt, m0, 32, tid);
  a2r = stage_load(x_lab, m0, 32, tid);

  v16bf cbh0 = load_bfrag(bh0, 0, hv), cbl0 = load_bfrag(bl0, 0, hv);
  v16bf cbh1 = load_bfrag(bh1, 0, hv), cbl1 = load_bfrag(bl1, 0, hv);
  v16bf cbh2 = load_bfrag(bh2, 0, hv), cbl2 = load_bfrag(bl2, 0, hv);
  __syncthreads();

#pragma unroll 2   // even/odd iterations swap buffer roles by renaming
  for (int kk = 0; kk < NK; ++kk) {
    const int k     = kk * 32;
    const int kn    = (k + 32) & (KDIM - 1);  // chunk kk+1 (wraps harmlessly)
    const int k2    = (k + 64) & (KDIM - 1);  // chunk kk+2
    const int bufB  = (kk & 1) * BUFB;
    const int nbufB = BUFB - bufB;

    // 1) regs (chunk kk+1) -> other LDS buffer
    stage_store(a0r, smem, nbufB + 0 * MODB, tid);
    stage_store(a1r, smem, nbufB + 1 * MODB, tid);
    stage_store(a2r, smem, nbufB + 2 * MODB, tid);

    // 2) issue chunk kk+2 global A loads (consumed next iteration)
    a0r = stage_load(x_ct,  m0, k2, tid);
    a1r = stage_load(x_txt, m0, k2, tid);
    a2r = stage_load(x_lab, m0, k2, tid);

    // 3) issue chunk kk+1 B loads (consumed next iteration)
    const v16bf nbh0 = load_bfrag(bh0, kn, hv);
    const v16bf nbl0 = load_bfrag(bl0, kn, hv);
    const v16bf nbh1 = load_bfrag(bh1, kn, hv);
    const v16bf nbl1 = load_bfrag(bl1, kn, hv);
    const v16bf nbh2 = load_bfrag(bh2, kn, hv);
    const v16bf nbl2 = load_bfrag(bl2, kn, hv);

    // 4) consume chunk kk: A fragments from LDS + WMMA with current B regs
    {
      const v16bf ah = load_afrag(smem, bufB + 0 * MODB, r16, hv);
      const v16bf al = load_afrag(smem, bufB + 0 * MODB + TILEB, r16, hv);
      acc0 = wmma3(ah, al, cbh0, cbl0, acc0);
    }
    {
      const v16bf ah = load_afrag(smem, bufB + 1 * MODB, r16, hv);
      const v16bf al = load_afrag(smem, bufB + 1 * MODB + TILEB, r16, hv);
      acc1 = wmma3(ah, al, cbh1, cbl1, acc1);
    }
    {
      const v16bf ah = load_afrag(smem, bufB + 2 * MODB, r16, hv);
      const v16bf al = load_afrag(smem, bufB + 2 * MODB + TILEB, r16, hv);
      acc2 = wmma3(ah, al, cbh2, cbl2, acc2);
    }

    __syncthreads();
    cbh0 = nbh0; cbl0 = nbl0;
    cbh1 = nbh1; cbl1 = nbl1;
    cbh2 = nbh2; cbl2 = nbl2;
  }

  // C/D layout: VGPR r -> row M = r + 8*hv, column N = lane & 15.
  float lsum = 0.0f;
#pragma unroll
  for (int r = 0; r < 8; ++r) {
    const int mo = m0 + 8 * hv + r;
    const int no = wave * 16 + r16;
    const float zc = acc0[r], zt = acc1[r], zl = acc2[r];
    aligned[(size_t)mo * NDIM + no] = (zc + zt + zl) * (1.0f / 3.0f);
    const float d0 = zc - zt, d1 = zc - zl, d2 = zt - zl;
    lsum += d0 * d0 + d1 * d1 + d2 * d2;
  }

  __shared__ float red[256];
  red[tid] = lsum;
  __syncthreads();
#pragma unroll
  for (int s = 128; s > 0; s >>= 1) {
    if (tid < s) red[tid] += red[tid + s];
    __syncthreads();
  }
  if (tid == 0) partials[blockIdx.x] = red[0];
}

__global__ __launch_bounds__(256) void
ot_finalize(const float* __restrict__ partials, float* __restrict__ loss_out) {
  __shared__ float red[256];
  red[threadIdx.x] = partials[threadIdx.x];
  __syncthreads();
#pragma unroll
  for (int s = 128; s > 0; s >>= 1) {
    if (threadIdx.x < s) red[threadIdx.x] += red[threadIdx.x + s];
    __syncthreads();
  }
  if (threadIdx.x == 0)
    *loss_out = red[0] / (3.0f * (float)BATCH_) - EPS_;
}

extern "C" void kernel_launch(void* const* d_in, const int* in_sizes, int n_in,
                              void* d_out, int out_size, void* d_ws,
                              size_t ws_size, hipStream_t stream) {
  (void)in_sizes; (void)n_in; (void)out_size; (void)ws_size;
  const float* x_ct  = (const float*)d_in[0];
  const float* x_txt = (const float*)d_in[1];
  const float* x_lab = (const float*)d_in[2];
  const float* W_ct  = (const float*)d_in[3];
  const float* W_txt = (const float*)d_in[4];
  const float* W_lab = (const float*)d_in[5];

  float* out = (float*)d_out;   // [4096*128] aligned, then [1] ot_loss

  // workspace layout: Whi[3*NW] bf16 | Wlo[3*NW] bf16 | partials[256] f32
  __bf16* Whi = (__bf16*)d_ws;
  __bf16* Wlo = Whi + 3 * (size_t)NW;
  float* partials = (float*)(Wlo + 3 * (size_t)NW);

  convert_weights<<<dim3(512, 3), dim3(256), 0, stream>>>(W_ct, W_txt, W_lab,
                                                          Whi, Wlo);
  ot_fused_gemm<<<dim3(BATCH_ / 16), dim3(256), 0, stream>>>(
      x_ct, x_txt, x_lab, Whi, Wlo, out, partials);
  ot_finalize<<<dim3(1), dim3(256), 0, stream>>>(partials,
                                                 out + (size_t)BATCH_ * NDIM);
}
